// HgnnEncoder_27264452395320
// MI455X (gfx1250) — matrix operand order repaired and checked by reading, not compiled, not run on hardware
//
#include <hip/hip_runtime.h>
#include <hip/hip_bf16.h>

// ---------------------------------------------------------------------------
// Problem constants (match reference)
// ---------------------------------------------------------------------------
#define N_NODES 50000
#define N_EDGES 10000
#define NNZ     1600000
#define IN_CH   256
#define DIM1    128
#define HEADS   4
#define FEAT    (HEADS * DIM1)   // 512
#define LN_EPS  1e-5f

// ---------------------------------------------------------------------------
// WMMA types
// ---------------------------------------------------------------------------
typedef __attribute__((ext_vector_type(16))) __bf16 v16bf;
typedef __attribute__((ext_vector_type(8)))  float  v8f;

union FragBF { uint4 q[2]; v16bf v; };

__device__ __forceinline__ unsigned short f2bf(float f) {
    unsigned int u = __float_as_uint(f);
    unsigned int r = u + 0x7FFFu + ((u >> 16) & 1u);   // round-to-nearest-even
    return (unsigned short)(r >> 16);
}

// ---------------------------------------------------------------------------
// Workspace layout (bytes)
// ---------------------------------------------------------------------------
#define OFF_XB   ((size_t)0)                                   // bf16 x   [N,256]
#define OFF_WT   (OFF_XB  + (size_t)N_NODES * IN_CH * 2)       // bf16 W^T [H,128,256]
#define OFF_XT   (OFF_WT  + (size_t)HEADS * DIM1 * IN_CH * 2)  // f32 xt   [H,N,128]
#define OFF_ME   (OFF_XT  + (size_t)HEADS * N_NODES * DIM1 * 4)// f32 m_e  [H,E,128]
#define OFF_DEG  (OFF_ME  + (size_t)HEADS * N_EDGES * DIM1 * 4)// f32 deg_e[E]
#define OFF_D    (OFF_DEG + (size_t)N_EDGES * 4)               // f32 D    [N]
#define ACC_FLOATS ((size_t)HEADS * N_EDGES * DIM1 + N_EDGES + N_NODES)

// ---------------------------------------------------------------------------
// 0) zero fill
// ---------------------------------------------------------------------------
__global__ void zero_f32(float* __restrict__ p, long n) {
    long i = (long)blockIdx.x * blockDim.x + threadIdx.x;
    long stride = (long)gridDim.x * blockDim.x;
    for (; i < n; i += stride) p[i] = 0.0f;
}

// ---------------------------------------------------------------------------
// 1) convert x (f32 row-major) -> bf16 row-major
// ---------------------------------------------------------------------------
__global__ void cvt_x_bf16(const float* __restrict__ x, unsigned short* __restrict__ xb) {
    long i = (long)blockIdx.x * blockDim.x + threadIdx.x;
    long n = (long)N_NODES * IN_CH;
    long stride = (long)gridDim.x * blockDim.x;
    for (; i < n; i += stride) xb[i] = f2bf(x[i]);
}

// ---------------------------------------------------------------------------
// 2) convert W [H,256,128] f32 -> W^T [H,128,256] bf16 (N-major => B fragments
//    become contiguous 32B loads per lane)
// ---------------------------------------------------------------------------
__global__ void cvt_w_bf16(const float* __restrict__ W, unsigned short* __restrict__ wt) {
    long i = (long)blockIdx.x * blockDim.x + threadIdx.x;
    long n = (long)HEADS * DIM1 * IN_CH;
    if (i >= n) return;
    int h = (int)(i / (DIM1 * IN_CH));
    int r = (int)(i % (DIM1 * IN_CH));
    int nn = r / IN_CH;      // output col
    int k  = r % IN_CH;      // input ch
    wt[i] = f2bf(W[(long)h * IN_CH * DIM1 + (long)k * DIM1 + nn]);
}

// ---------------------------------------------------------------------------
// 3) WMMA GEMM: xt[h] = x @ W[h]   (bf16 x bf16 -> f32)
//    one wave per 16x16 output tile; K=256 in 8 steps of 32.
//    A layout (16-bit 16x32): lane L (M=L%16), elems j: K = (j/8)*16 + (L>=16?8:0) + j%8
//    B layout (16-bit 32x16): lane L (N=L%16), elems j: K = (j/8)*16 + (L>=16?8:0) + j%8
//    C/D: vgpr r, lane L: M = r + (L>=16?8:0), N = L%16
// ---------------------------------------------------------------------------
__global__ void gemm_xt_wmma(const unsigned short* __restrict__ xb,
                             const unsigned short* __restrict__ wt,
                             float* __restrict__ xt) {
    const int lane = threadIdx.x & 31;
    const int wave = threadIdx.x >> 5;
    const long MT = N_NODES / 16;                       // 3125
    const long NT = DIM1 / 16;                          // 8
    const long task = (long)blockIdx.x * 8 + wave;      // MT*NT*HEADS = 100000 tasks
    if (task >= MT * NT * HEADS) return;                // wave-uniform: EXEC stays full

    const int head = (int)(task / (MT * NT));
    const long rem = task % (MT * NT);
    const int mt = (int)(rem / NT);
    const int nt = (int)(rem % NT);
    const int m0 = mt * 16, n0 = nt * 16;
    const int half = lane >> 4, lm = lane & 15;

    const unsigned short* arow = xb + (long)(m0 + lm) * IN_CH;
    const unsigned short* bcol = wt + ((long)head * DIM1 + n0 + lm) * IN_CH;

    v8f c = {};
#pragma unroll
    for (int ks = 0; ks < 8; ++ks) {
        const int kb = ks * 32;
        FragBF a, b;
        a.q[0] = *(const uint4*)(arow + kb + half * 8);
        a.q[1] = *(const uint4*)(arow + kb + 16 + half * 8);
        b.q[0] = *(const uint4*)(bcol + kb + half * 8);
        b.q[1] = *(const uint4*)(bcol + kb + 16 + half * 8);
        c = __builtin_amdgcn_wmma_f32_16x16x32_bf16(
                false, a.v, false, b.v, (short)0, c, false, false);
    }

    float* out = xt + ((long)head * N_NODES + m0) * DIM1 + n0 + lm;
#pragma unroll
    for (int r = 0; r < 8; ++r)
        out[(long)(r + half * 8) * DIM1] = c[r];
}

// ---------------------------------------------------------------------------
// 4) degrees: deg_e[e] += 1 ; D[n] += edge_w[e]
// ---------------------------------------------------------------------------
__global__ void degrees(const int* __restrict__ ni, const int* __restrict__ ei,
                        const float* __restrict__ ew,
                        float* __restrict__ deg, float* __restrict__ Dn) {
    long i = (long)blockIdx.x * blockDim.x + threadIdx.x;
    if (i >= NNZ) return;
    int e = ei[i];
    atomicAdd(&deg[e], 1.0f);
    atomicAdd(&Dn[ni[i]], ew[e]);
}

// ---------------------------------------------------------------------------
// 5) node -> edge scatter: m_e[h][e] += xt[h][n]   (wave per nnz, lane = 4 feats)
// ---------------------------------------------------------------------------
__global__ void scatter_n2e(const int* __restrict__ ni, const int* __restrict__ ei,
                            const float* __restrict__ xt, float* __restrict__ me) {
    const int lane = threadIdx.x & 31;
    const long w = (long)blockIdx.x * 8 + (threadIdx.x >> 5);
    if (w >= NNZ) return;
    const int n = ni[w], e = ei[w];
#pragma unroll
    for (int h = 0; h < HEADS; ++h) {
        const float4 v = *(const float4*)(xt + ((long)h * N_NODES + n) * DIM1 + lane * 4);
        float* dst = me + ((long)h * N_EDGES + e) * DIM1 + lane * 4;
        atomicAdd(dst + 0, v.x);
        atomicAdd(dst + 1, v.y);
        atomicAdd(dst + 2, v.z);
        atomicAdd(dst + 3, v.w);
    }
}

// ---------------------------------------------------------------------------
// 6) fold: m_e *= (edge_w[e] / deg_e[e])   (Binv and per-edge weight together)
// ---------------------------------------------------------------------------
__global__ void scale_me(float* __restrict__ me, const float* __restrict__ deg,
                         const float* __restrict__ ew) {
    long i = (long)blockIdx.x * blockDim.x + threadIdx.x;
    long n = (long)HEADS * N_EDGES * DIM1;
    if (i >= n) return;
    int e = (int)((i >> 7) % N_EDGES);
    float d = deg[e];
    float s = d > 0.0f ? ew[e] / d : 0.0f;
    me[i] *= s;
}

// ---------------------------------------------------------------------------
// 7) edge -> node scatter: out[n][h*128+d] += me_w[h][e][d]
// ---------------------------------------------------------------------------
__global__ void scatter_e2n(const int* __restrict__ ni, const int* __restrict__ ei,
                            const float* __restrict__ me, float* __restrict__ out) {
    const int lane = threadIdx.x & 31;
    const long w = (long)blockIdx.x * 8 + (threadIdx.x >> 5);
    if (w >= NNZ) return;
    const int n = ni[w], e = ei[w];
#pragma unroll
    for (int h = 0; h < HEADS; ++h) {
        const float4 v = *(const float4*)(me + ((long)h * N_EDGES + e) * DIM1 + lane * 4);
        float* dst = out + (long)n * FEAT + h * DIM1 + lane * 4;
        atomicAdd(dst + 0, v.x);
        atomicAdd(dst + 1, v.y);
        atomicAdd(dst + 2, v.z);
        atomicAdd(dst + 3, v.w);
    }
}

// ---------------------------------------------------------------------------
// 8) finalize: val = raw*Dinv + b ; relu ; LayerNorm(512) * gamma + beta
//    one wave32 per node: 16 feats/lane, shfl_xor tree reduce for mean/var
// ---------------------------------------------------------------------------
__global__ void finalize_ln(float* __restrict__ out, const float* __restrict__ Dn,
                            const float* __restrict__ bias,
                            const float* __restrict__ gamma,
                            const float* __restrict__ beta) {
    const int lane = threadIdx.x & 31;
    const long node = (long)blockIdx.x * 8 + (threadIdx.x >> 5);
    if (node >= N_NODES) return;
    const float d = Dn[node];
    const float dinv = d > 0.0f ? 1.0f / d : 0.0f;
    float* row = out + node * (long)FEAT;

    float vals[16];
    float s = 0.0f, ss = 0.0f;
#pragma unroll
    for (int q = 0; q < 4; ++q) {
        const int f = lane * 16 + q * 4;
        const float4 v  = *(const float4*)(row + f);
        const float4 bb = *(const float4*)(bias + f);
        float t;
        t = fmaxf(v.x * dinv + bb.x, 0.0f); vals[q * 4 + 0] = t; s += t; ss += t * t;
        t = fmaxf(v.y * dinv + bb.y, 0.0f); vals[q * 4 + 1] = t; s += t; ss += t * t;
        t = fmaxf(v.z * dinv + bb.z, 0.0f); vals[q * 4 + 2] = t; s += t; ss += t * t;
        t = fmaxf(v.w * dinv + bb.w, 0.0f); vals[q * 4 + 3] = t; s += t; ss += t * t;
    }
#pragma unroll
    for (int m = 16; m >= 1; m >>= 1) {
        s  += __shfl_xor(s,  m, 32);
        ss += __shfl_xor(ss, m, 32);
    }
    const float mean = s * (1.0f / FEAT);
    const float var  = ss * (1.0f / FEAT) - mean * mean;
    const float inv  = rsqrtf(var + LN_EPS);
#pragma unroll
    for (int q = 0; q < 4; ++q) {
        const int f = lane * 16 + q * 4;
        const float4 g = *(const float4*)(gamma + f);
        const float4 bt = *(const float4*)(beta + f);
        float4 o;
        o.x = (vals[q * 4 + 0] - mean) * inv * g.x + bt.x;
        o.y = (vals[q * 4 + 1] - mean) * inv * g.y + bt.y;
        o.z = (vals[q * 4 + 2] - mean) * inv * g.z + bt.z;
        o.w = (vals[q * 4 + 3] - mean) * inv * g.w + bt.w;
        *(float4*)(row + f) = o;
    }
}

// ---------------------------------------------------------------------------
// launch
// ---------------------------------------------------------------------------
extern "C" void kernel_launch(void* const* d_in, const int* in_sizes, int n_in,
                              void* d_out, int out_size, void* d_ws, size_t ws_size,
                              hipStream_t stream) {
    const float* x     = (const float*)d_in[0];
    const float* W     = (const float*)d_in[1];
    const float* bias  = (const float*)d_in[2];
    const float* gamma = (const float*)d_in[3];
    const float* beta  = (const float*)d_in[4];
    const int*   edge  = (const int*)d_in[5];
    const float* ew    = (const float*)d_in[6];
    (void)d_in; (void)in_sizes; (void)n_in; (void)out_size; (void)ws_size;

    const int* node_idx = edge;
    const int* edge_idx = edge + NNZ;

    char* ws = (char*)d_ws;
    unsigned short* xb  = (unsigned short*)(ws + OFF_XB);
    unsigned short* wt  = (unsigned short*)(ws + OFF_WT);
    float*          xt  = (float*)(ws + OFF_XT);
    float*          me  = (float*)(ws + OFF_ME);
    float*          deg = (float*)(ws + OFF_DEG);
    float*          Dn  = (float*)(ws + OFF_D);
    float*          out = (float*)d_out;

    // 0) zero accumulators (m_e, deg_e, D are contiguous) and d_out
    {
        long nacc = (long)ACC_FLOATS;
        zero_f32<<<dim3(20240), dim3(256), 0, stream>>>(me, nacc);
        long nout = (long)N_NODES * FEAT;
        zero_f32<<<dim3(100000), dim3(256), 0, stream>>>(out, nout);
    }
    // 1) x -> bf16
    cvt_x_bf16<<<dim3(50000), dim3(256), 0, stream>>>(x, xb);
    // 2) W -> bf16 transposed
    cvt_w_bf16<<<dim3((HEADS * DIM1 * IN_CH + 255) / 256), dim3(256), 0, stream>>>(W, wt);
    // 3) WMMA GEMM (100000 wave-tasks, 8 waves/block)
    gemm_xt_wmma<<<dim3(12500), dim3(256), 0, stream>>>(xb, wt, xt);
    // 4) degrees
    degrees<<<dim3((NNZ + 255) / 256), dim3(256), 0, stream>>>(node_idx, edge_idx, ew, deg, Dn);
    // 5) node -> edge scatter (wave per nnz)
    scatter_n2e<<<dim3(NNZ / 8), dim3(256), 0, stream>>>(node_idx, edge_idx, xt, me);
    // 6) scale m_e by edge_w / deg
    scale_me<<<dim3((HEADS * N_EDGES * DIM1 + 255) / 256), dim3(256), 0, stream>>>(me, deg, ew);
    // 7) edge -> node scatter
    scatter_e2n<<<dim3(NNZ / 8), dim3(256), 0, stream>>>(node_idx, edge_idx, me, out);
    // 8) bias + relu + layernorm (wave per node)
    finalize_ln<<<dim3((N_NODES + 7) / 8), dim3(256), 0, stream>>>(out, Dn, bias, gamma, beta);
}
